// SeqAttnMatchV1_63694365000418
// MI455X (gfx1250) — matrix operand order repaired and checked by reading, not compiled, not run on hardware
//
#include <hip/hip_runtime.h>
#include <hip/hip_bf16.h>
#include <math.h>

typedef __attribute__((ext_vector_type(16))) _Float16 v16h;
typedef __attribute__((ext_vector_type(8)))  _Float16 v8h;
typedef __attribute__((ext_vector_type(4)))  _Float16 v4h;
typedef __attribute__((ext_vector_type(8)))  float    v8f;

#define BATCH 4096
#define HLEN  200
#define DIM   256
#define MPAD  208      // 13 M-tiles of 16 (200 history rows + tweet row + pad)
#define MT    13
#define XSTR  72       // X tile row stride in halves (64 + 8 pad, 16B aligned)
#define PSTR  264      // P buffer row stride in halves (256 + 8 pad)

union V16U { v16h v; v8h h[2]; };

__global__ void convert_W_f16(const float* __restrict__ W, _Float16* __restrict__ W16) {
    int i = blockIdx.x * blockDim.x + threadIdx.x;
    W16[i] = (_Float16)W[i];
}

__global__ __launch_bounds__(512, 1)
void seq_attn_match_kernel(const float* __restrict__ tweet,
                           const float* __restrict__ history,
                           const unsigned char* __restrict__ mask,
                           const _Float16* __restrict__ W16,
                           const float* __restrict__ bias,
                           float* __restrict__ out) {
    __shared__ _Float16 Xbuf[MPAD * XSTR];   // 29,952 B : f16 X chunk [208 x 64]
    __shared__ _Float16 Pbuf[MPAD * PSTR];   // 109,824 B: f16 relu(XW^T+b) [208 x 256]
    __shared__ float scoresS[HLEN];
    __shared__ float alphaS[HLEN];
    __shared__ float red0;

    const int tid  = threadIdx.x;
    const int w    = tid >> 5;          // wave id 0..15, owns N-tile w
    const int lane = tid & 31;
    const int b    = blockIdx.x;
    const int hi   = (lane >= 16) ? 1 : 0;
    const int l16  = lane & 15;

    v8f acc[MT];
    const v8f zero = {0.f, 0.f, 0.f, 0.f, 0.f, 0.f, 0.f, 0.f};
#pragma unroll
    for (int m = 0; m < MT; ++m) acc[m] = zero;

    const float bval = bias[w * 16 + l16];

    // A-fragment gather from LDS (16x32 f16 layout):
    // lanes 0-15: K 0..7 / 16..23, lanes 16-31: K 8..15 / 24..31
    auto loadA = [&](int ks, int m) {
        const _Float16* xp = Xbuf + (m * 16 + l16) * XSTR + ks * 32 + hi * 8;
        V16U af;
        af.h[0] = *(const v8h*)(xp);
        af.h[1] = *(const v8h*)(xp + 16);
        return af;
    };

    // ---------------- Phase 1: fused GEMM  P = relu([history[b]; tweet[b]] @ W^T + b)
    for (int kc = 0; kc < 4; ++kc) {
        const int k0 = kc * 64;
        // cooperative f32->f16 load of X chunk (rows 0..199 history, 200 tweet, 201..207 zero)
        for (int slot = tid; slot < MPAD * 16; slot += 512) {
            const int row = slot >> 4;
            const int kq  = (slot & 15) << 2;
            float4 v = make_float4(0.f, 0.f, 0.f, 0.f);
            if (row < HLEN)
                v = *(const float4*)(history + ((size_t)b * HLEN + row) * DIM + k0 + kq);
            else if (row == HLEN)
                v = *(const float4*)(tweet + (size_t)b * DIM + k0 + kq);
            v4h h;
            h.x = (_Float16)v.x; h.y = (_Float16)v.y;
            h.z = (_Float16)v.z; h.w = (_Float16)v.w;
            *(v4h*)(Xbuf + row * XSTR + kq) = h;
        }
        __syncthreads();

        // B fragments (32x16): lane = column w*16+l16; K 8h..8h+7 / 16+8h..16+8h+7 of W row
        const _Float16* wp0 = W16 + (size_t)(w * 16 + l16) * DIM + k0 + hi * 8;
        V16U bf0, bf1;
        bf0.h[0] = *(const v8h*)(wp0);
        bf0.h[1] = *(const v8h*)(wp0 + 16);
        bf1.h[0] = *(const v8h*)(wp0 + 32);
        bf1.h[1] = *(const v8h*)(wp0 + 48);

        // software-pipelined over flattened (ks, m): prefetch A frag one step ahead
        V16U af = loadA(0, 0);
#pragma unroll
        for (int step = 0; step < 2 * MT; ++step) {
            const int ks = step / MT;
            const int m  = step % MT;
            V16U afn;
            if (step + 1 < 2 * MT)
                afn = loadA((step + 1) / MT, (step + 1) % MT);
            else
                afn = af;
            acc[m] = __builtin_amdgcn_wmma_f32_16x16x32_f16(
                false, af.v, false, (ks ? bf1.v : bf0.v), (short)0, acc[m],
                false, false);
            af = afn;
        }

        // Scheduler shape for this region: 52 DS reads, 26 WMMA, 4 VMEM reads.
        // Prefetch 2 A-fragments (4 ds_load_b128), then alternate
        // {1 WMMA, next fragment's 2 ds_load_b128} x24, then the last 2 WMMAs.
        __builtin_amdgcn_sched_group_barrier(0x020, 4, 0);  // 4 global B-frag loads first
        __builtin_amdgcn_sched_group_barrier(0x100, 4, 0);  // A-frags 0,1
#pragma unroll
        for (int i = 0; i < 24; ++i) {
            __builtin_amdgcn_sched_group_barrier(0x408, 1, 0);  // WMMA i
            __builtin_amdgcn_sched_group_barrier(0x100, 2, 0);  // A-frag i+2
        }
        __builtin_amdgcn_sched_group_barrier(0x408, 1, 0);      // WMMA 24
        __builtin_amdgcn_sched_group_barrier(0x408, 1, 0);      // WMMA 25

        __syncthreads();
    }

    // epilogue: bias + relu, store f16 into Pbuf
#pragma unroll
    for (int m = 0; m < MT; ++m) {
#pragma unroll
        for (int r = 0; r < 8; ++r) {
            float x = acc[m][r] + bval;
            x = fmaxf(x, 0.0f);
            const int row = m * 16 + r + hi * 8;   // C/D layout: VGPR r -> row r / r+8
            Pbuf[row * PSTR + w * 16 + l16] = (_Float16)x;
        }
    }
    __syncthreads();

    // ---------------- Phase 2: scores[l] = tweet_proj . hist_proj[l], masked
    for (int l = w; l < HLEN; l += 16) {
        float s = 0.0f;
        for (int d = lane; d < DIM; d += 32)
            s += (float)Pbuf[HLEN * PSTR + d] * (float)Pbuf[l * PSTR + d];
#pragma unroll
        for (int off = 16; off > 0; off >>= 1)
            s += __shfl_xor(s, off, 32);
        if (lane == 0)
            scoresS[l] = mask[(size_t)b * HLEN + l] ? -INFINITY : s;
    }
    __syncthreads();

    // softmax reduction on wave 0
    if (w == 0) {
        float mx = -INFINITY;
        for (int l = lane; l < HLEN; l += 32) mx = fmaxf(mx, scoresS[l]);
#pragma unroll
        for (int off = 16; off > 0; off >>= 1)
            mx = fmaxf(mx, __shfl_xor(mx, off, 32));
        float sum = 0.0f;
        for (int l = lane; l < HLEN; l += 32) {
            float e = __expf(scoresS[l] - mx);
            alphaS[l] = e;
            sum += e;
        }
#pragma unroll
        for (int off = 16; off > 0; off >>= 1)
            sum += __shfl_xor(sum, off, 32);
        if (lane == 0) red0 = 1.0f / sum;
    }
    __syncthreads();

    // ---------------- Phase 3: matched_seq[d] = sum_l alpha[l] * hist_proj[l][d]
    if (tid < DIM) {
        const float inv = red0;
        float a = 0.0f;
        for (int l = 0; l < HLEN; ++l)
            a += alphaS[l] * (float)Pbuf[l * PSTR + tid];
        out[(size_t)b * DIM + tid] = a * inv;
    }
}

extern "C" void kernel_launch(void* const* d_in, const int* in_sizes, int n_in,
                              void* d_out, int out_size, void* d_ws, size_t ws_size,
                              hipStream_t stream) {
    (void)in_sizes; (void)n_in; (void)out_size; (void)ws_size;
    const float*         tweet   = (const float*)d_in[0];
    const float*         history = (const float*)d_in[1];
    const unsigned char* maskp   = (const unsigned char*)d_in[2];
    const float*         W       = (const float*)d_in[3];
    const float*         bias    = (const float*)d_in[4];
    float*    out = (float*)d_out;
    _Float16* W16 = (_Float16*)d_ws;   // 128 KB scratch for f16 weights

    convert_W_f16<<<(DIM * DIM) / 256, 256, 0, stream>>>(W, W16);
    seq_attn_match_kernel<<<BATCH, 512, 0, stream>>>(tweet, history, maskp, W16, bias, out);
}
